// Model_70136815943965
// MI455X (gfx1250) — matrix operand order repaired and checked by reading, not compiled, not run on hardware
//
#include <hip/hip_runtime.h>
#include <hip/hip_bf16.h>

// ---------------- model dims ----------------
#define NB    16          // batch
#define NC    32          // vars
#define NLP   32          // patches per var (512/16)
#define NTOK  16384       // NB*NC*NLP tokens
#define LSEQ  1024        // tokens per batch (NC*NLP)
#define DM    128         // d_model
#define DI    256         // d_inner
#define NST   16          // d_state
#define DTR   8           // dt_rank
#define XDB   48          // dt_rank + 2*d_state (40) padded to 48
#define PRED  96
#define KHEAD 4096        // DM * NLP

typedef __attribute__((ext_vector_type(16))) _Float16 v16h;
typedef __attribute__((ext_vector_type(8)))  _Float16 v8h;
typedef __attribute__((ext_vector_type(8)))  float    v8f;

union FragH { v16h v; _Float16 h[16]; v8h q[2]; };
union FragF { v8f  v; float f[8]; };

// ---------------- instance norm + patchify into f16 A (K padded 16->32) ----------
__global__ void __launch_bounds__(256) k_instnorm(
    const float* __restrict__ xe,     // (NB, 512, NC)
    float* __restrict__ meanv, float* __restrict__ stdv,   // (NB*NC)
    _Float16* __restrict__ Ap)        // (NTOK, 32)
{
  int bc = blockIdx.x;                // 0..511
  int b = bc >> 5, c = bc & 31;
  int tid = threadIdx.x;              // 256 threads, 2 time steps each
  int t0 = 2 * tid;
  float x0 = xe[((size_t)b * 512 + t0) * NC + c];
  float x1 = xe[((size_t)b * 512 + t0 + 1) * NC + c];
  float s = x0 + x1, sq = x0 * x0 + x1 * x1;
  for (int off = 16; off > 0; off >>= 1) {
    s  += __shfl_down(s,  off, 32);
    sq += __shfl_down(sq, off, 32);
  }
  __shared__ float ps[8], pq[8];
  __shared__ float sm, sisd;
  if ((tid & 31) == 0) { ps[tid >> 5] = s; pq[tid >> 5] = sq; }
  __syncthreads();
  if (tid == 0) {
    float S = 0.f, Q = 0.f;
    for (int i = 0; i < 8; i++) { S += ps[i]; Q += pq[i]; }
    float m  = S / 512.f;
    float sd = sqrtf(Q / 512.f - m * m + 1e-5f);
    meanv[bc] = m; stdv[bc] = sd;
    sm = m; sisd = 1.f / sd;
  }
  __syncthreads();
  float m = sm, isd = sisd;
  {
    int lp = t0 >> 4, k = t0 & 15;
    Ap[((size_t)bc * NLP + lp) * 32 + k] = (_Float16)((x0 - m) * isd);
  }
  {
    int t1 = t0 + 1, lp = t1 >> 4, k = t1 & 15;
    Ap[((size_t)bc * NLP + lp) * 32 + k] = (_Float16)((x1 - m) * isd);
  }
  // zero K-pad (32 tokens * 16 halves = 512 slots, 2 per thread)
  for (int j = 0; j < 2; j++) {
    int p = 2 * tid + j;
    int lp = p >> 4, k = p & 15;
    Ap[((size_t)bc * NLP + lp) * 32 + 16 + k] = (_Float16)0.f;
  }
}

// ---------------- f32 -> f16 weight convert with row/K zero-pad ----------------
__global__ void k_cvt(const float* __restrict__ src, _Float16* __restrict__ dst,
                      int rowsDst, int rowsSrc, int Ksrc, int Kdst)
{
  long i = (long)blockIdx.x * blockDim.x + threadIdx.x;
  long total = (long)rowsDst * Kdst;
  if (i >= total) return;
  int r = (int)(i / Kdst), k = (int)(i % Kdst);
  float v = (r < rowsSrc && k < Ksrc) ? src[(size_t)r * Ksrc + k] : 0.f;
  dst[i] = (_Float16)v;
}

// ---------------- WMMA GEMM: C[M,N](f32) = A[M,K](f16) * W[N,K](f16)^T (+R) ----
// Each wave computes a 64x16 C macro-tile: the W fragment is loaded once per
// K-chunk and reused by 4 WMMAs (4 A fragments) -> 4 wmma per 10 b128 loads.
// Requires M % 64 == 0, N % 16 == 0, K % 32 == 0 (true for every call here).
__global__ void __launch_bounds__(256) k_gemm_f16(
    const _Float16* __restrict__ A, const _Float16* __restrict__ W,
    const float* __restrict__ R, float* __restrict__ C,
    int M, int N, int K)
{
  const int lane  = threadIdx.x & 31;
  const int wave  = blockIdx.x * (blockDim.x >> 5) + (threadIdx.x >> 5);
  const int nTiles = N >> 4;
  const int m64   = wave / nTiles;
  const int nTile = wave - m64 * nTiles;
  if (m64 * 64 >= M) return;
  const int r15 = lane & 15;
  const int hi  = lane >> 4;
  // A fragment (ISA 16-bit A 16x32 layout): lane holds row r15,
  // halves 0..7 = K kb..kb+7, halves 8..15 = K kb+16..kb+23, kb = 8*hi
  const _Float16* aPtr = A + (size_t)(m64 * 64 + r15) * K + (hi ? 8 : 0);
  // B fragment: lane holds column r15 (= row of W), same K layout as A
  const _Float16* wPtr = W + (size_t)(nTile * 16 + r15) * K + (hi ? 8 : 0);
  const size_t aStep = (size_t)16 * K;     // 16 rows
  v8f acc0 = {}, acc1 = {}, acc2 = {}, acc3 = {};
  for (int k = 0; k < K; k += 32) {
    if (k + 64 < K) {
      __builtin_prefetch(aPtr + k + 64, 0, 0);
      __builtin_prefetch(wPtr + k + 64, 0, 0);
    }
    FragH b;
    b.q[0] = *(const v8h*)(wPtr + k);
    b.q[1] = *(const v8h*)(wPtr + k + 16);
    FragH a0, a1, a2, a3;
    a0.q[0] = *(const v8h*)(aPtr + k);
    a0.q[1] = *(const v8h*)(aPtr + k + 16);
    a1.q[0] = *(const v8h*)(aPtr + aStep + k);
    a1.q[1] = *(const v8h*)(aPtr + aStep + k + 16);
    a2.q[0] = *(const v8h*)(aPtr + 2 * aStep + k);
    a2.q[1] = *(const v8h*)(aPtr + 2 * aStep + k + 16);
    a3.q[0] = *(const v8h*)(aPtr + 3 * aStep + k);
    a3.q[1] = *(const v8h*)(aPtr + 3 * aStep + k + 16);
    acc0 = __builtin_amdgcn_wmma_f32_16x16x32_f16(
        false, a0.v, false, b.v, (short)0, acc0, false, false);
    acc1 = __builtin_amdgcn_wmma_f32_16x16x32_f16(
        false, a1.v, false, b.v, (short)0, acc1, false, false);
    acc2 = __builtin_amdgcn_wmma_f32_16x16x32_f16(
        false, a2.v, false, b.v, (short)0, acc2, false, false);
    acc3 = __builtin_amdgcn_wmma_f32_16x16x32_f16(
        false, a3.v, false, b.v, (short)0, acc3, false, false);
  }
  const int n = nTile * 16 + r15;
  v8f accs[4] = { acc0, acc1, acc2, acc3 };
  #pragma unroll
  for (int i = 0; i < 4; i++) {
    FragF cf; cf.v = accs[i];
    const int mBase = m64 * 64 + i * 16 + (hi << 3);
    #pragma unroll
    for (int j = 0; j < 8; j++) {
      size_t idx = (size_t)(mBase + j) * N + n;
      float v = cf.f[j];
      if (R) v += R[idx];
      C[idx] = v;
    }
  }
}

// ---------------- RMSNorm over d_model, emit f16 ----------------
__global__ void __launch_bounds__(128) k_rmsnorm(
    const float* __restrict__ x, const float* __restrict__ w,
    _Float16* __restrict__ o)
{
  int t = blockIdx.x;         // token
  int d = threadIdx.x;        // 0..127
  float v = x[(size_t)t * DM + d];
  float ss = v * v;
  for (int off = 16; off > 0; off >>= 1) ss += __shfl_down(ss, off, 32);
  __shared__ float p[4];
  if ((d & 31) == 0) p[d >> 5] = ss;
  __syncthreads();
  float tot = p[0] + p[1] + p[2] + p[3];
  float r = rsqrtf(tot / (float)DM + 1e-5f);
  o[(size_t)t * DM + d] = (_Float16)(v * r * w[d]);
}

// ---------------- causal depthwise conv (k=4) + SiLU ----------------
__global__ void k_conv_silu(const float* __restrict__ xz,  // (NTOK, 2*DI), x = cols 0..255
                            const float* __restrict__ cw,  // (DI,4)
                            const float* __restrict__ cb,  // (DI)
                            float* __restrict__ u, _Float16* __restrict__ u16)
{
  int i = blockIdx.x * blockDim.x + threadIdx.x;
  if (i >= NTOK * DI) return;
  int di = i & (DI - 1);
  int tok = i >> 8;
  int tl = tok & (LSEQ - 1);
  int b  = tok >> 10;
  float s = cb[di];
  #pragma unroll
  for (int k = 0; k < 4; k++) {
    int t2 = tl - 3 + k;
    if (t2 >= 0)
      s += xz[((size_t)(b * LSEQ + t2)) * (2 * DI) + di] * cw[di * 4 + k];
  }
  float si = s / (1.f + __expf(-s));
  u[i] = si;
  u16[i] = (_Float16)si;
}

// ---------------- dt = softplus(dt_rank-proj + bias) ----------------
__global__ void k_dt(const float* __restrict__ xdbl,   // (NTOK, XDB), dt = cols 0..7
                     const float* __restrict__ dtw,    // (DI, DTR)
                     const float* __restrict__ dtbias, // (DI)
                     float* __restrict__ dtb)          // (NTOK, DI)
{
  int i = blockIdx.x * blockDim.x + threadIdx.x;
  if (i >= NTOK * DI) return;
  int di = i & (DI - 1);
  int tok = i >> 8;
  float s = dtbias[di];
  #pragma unroll
  for (int r = 0; r < DTR; r++)
    s += xdbl[(size_t)tok * XDB + r] * dtw[di * DTR + r];
  dtb[i] = (s > 20.f) ? s : log1pf(__expf(s));
}

// ---------------- selective scan (sequential over L=1024) ----------------
// one block per batch, one thread per d_inner channel, h[16] in registers,
// B/C vectors staged in LDS each step; fused D-skip + SiLU(z) gate, f16 out.
__global__ void __launch_bounds__(256) k_scan(
    const float* __restrict__ dtb,   // (NTOK, DI)
    const float* __restrict__ u,     // (NTOK, DI)
    const float* __restrict__ xdbl,  // (NTOK, XDB), B = 8..23, C = 24..39
    const float* __restrict__ xz,    // (NTOK, 2*DI), z = cols 256..511
    const float* __restrict__ Alog,  // (DI, NST)
    const float* __restrict__ Dp,    // (DI)
    _Float16* __restrict__ y16)      // (NTOK, DI)
{
  int b  = blockIdx.x;
  int di = threadIdx.x;
  float Arow[NST];
  #pragma unroll
  for (int n = 0; n < NST; n++) Arow[n] = -__expf(Alog[di * NST + n]);
  float dsk = Dp[di];
  float h[NST];
  #pragma unroll
  for (int n = 0; n < NST; n++) h[n] = 0.f;
  __shared__ float sBC[2 * NST];
  for (int t = 0; t < LSEQ; t++) {
    int tok = b * LSEQ + t;
    if (di < 2 * NST) sBC[di] = xdbl[(size_t)tok * XDB + DTR + di];
    __syncthreads();
    float dtv = dtb[(size_t)tok * DI + di];
    float uv  = u[(size_t)tok * DI + di];
    float y = uv * dsk;
    #pragma unroll
    for (int n = 0; n < NST; n++) {
      h[n] = __expf(dtv * Arow[n]) * h[n] + dtv * sBC[n] * uv;
      y += h[n] * sBC[NST + n];
    }
    float zv = xz[(size_t)tok * (2 * DI) + DI + di];
    float g  = zv / (1.f + __expf(-zv));
    y16[(size_t)tok * DI + di] = (_Float16)(y * g);
    __syncthreads();
  }
}

// ---------------- head epilogue: bias + transpose + de-normalize ----------------
__global__ void k_head_out(const float* __restrict__ ho,   // (NB*NC, PRED)
                           const float* __restrict__ hb,   // (PRED)
                           const float* __restrict__ meanv,
                           const float* __restrict__ stdv,
                           float* __restrict__ outp)       // (NB, PRED, NC)
{
  int i = blockIdx.x * blockDim.x + threadIdx.x;
  if (i >= NB * PRED * NC) return;
  int c = i & 31;
  int p = (i >> 5) % PRED;
  int b = i / (PRED * NC);
  int bc = b * NC + c;
  float v = ho[(size_t)bc * PRED + p] + hb[p];
  outp[i] = v * stdv[bc] + meanv[bc];
}

// ================================================================
extern "C" void kernel_launch(void* const* d_in, const int* in_sizes, int n_in,
                              void* d_out, int out_size, void* d_ws, size_t ws_size,
                              hipStream_t stream) {
  const float* x_enc        = (const float*)d_in[0];
  const float* patch_w      = (const float*)d_in[4];
  const float* norm_w       = (const float*)d_in[5];
  const float* in_proj_w    = (const float*)d_in[6];
  const float* conv_w       = (const float*)d_in[7];
  const float* conv_b       = (const float*)d_in[8];
  const float* x_proj_w     = (const float*)d_in[9];
  const float* dt_proj_w    = (const float*)d_in[10];
  const float* dt_proj_b    = (const float*)d_in[11];
  const float* A_log        = (const float*)d_in[12];
  const float* D_skip       = (const float*)d_in[13];
  const float* out_proj_w   = (const float*)d_in[14];
  const float* final_norm_w = (const float*)d_in[15];
  const float* head_w       = (const float*)d_in[16];
  const float* head_b       = (const float*)d_in[17];
  float* outp = (float*)d_out;

  // ---- carve workspace ----
  size_t off = 0;
  auto alloc = [&](size_t bytes) -> void* {
    void* p = (char*)d_ws + off;
    off = (off + bytes + 255) & ~(size_t)255;
    return p;
  };
  float*     meanv  = (float*)alloc(NB * NC * 4);
  float*     stdv   = (float*)alloc(NB * NC * 4);
  _Float16*  Ap     = (_Float16*)alloc((size_t)NTOK * 32 * 2);
  _Float16*  Wp     = (_Float16*)alloc((size_t)DM * 32 * 2);
  _Float16*  Win0   = (_Float16*)alloc((size_t)2 * DI * DM * 2);
  _Float16*  Win1   = (_Float16*)alloc((size_t)2 * DI * DM * 2);
  _Float16*  Wxp0   = (_Float16*)alloc((size_t)XDB * DI * 2);
  _Float16*  Wxp1   = (_Float16*)alloc((size_t)XDB * DI * 2);
  _Float16*  Wout0  = (_Float16*)alloc((size_t)DM * DI * 2);
  _Float16*  Wout1  = (_Float16*)alloc((size_t)DM * DI * 2);
  _Float16*  Whead  = (_Float16*)alloc((size_t)PRED * KHEAD * 2);
  float*     x      = (float*)alloc((size_t)NTOK * DM * 4);
  _Float16*  h16    = (_Float16*)alloc((size_t)NTOK * DM * 2);
  float*     xz     = (float*)alloc((size_t)NTOK * 2 * DI * 4);
  float*     u      = (float*)alloc((size_t)NTOK * DI * 4);
  _Float16*  u16    = (_Float16*)alloc((size_t)NTOK * DI * 2);
  float*     xdbl   = (float*)alloc((size_t)NTOK * XDB * 4);
  float*     dtb    = (float*)alloc((size_t)NTOK * DI * 4);
  _Float16*  y16    = (_Float16*)alloc((size_t)NTOK * DI * 2);
  float*     ho     = (float*)alloc((size_t)NB * NC * PRED * 4);
  (void)ws_size; (void)in_sizes; (void)n_in; (void)out_size;

  auto cvt = [&](const float* s, _Float16* d, int rD, int rS, int Ks, int Kd) {
    long total = (long)rD * Kd;
    k_cvt<<<(int)((total + 255) / 256), 256, 0, stream>>>(s, d, rD, rS, Ks, Kd);
  };
  auto gemm = [&](const _Float16* A, const _Float16* W, const float* R,
                  float* C, int M, int N, int K) {
    int waves = (M / 64) * (N / 16);        // 64x16 macro-tile per wave
    k_gemm_f16<<<(waves + 7) / 8, 256, 0, stream>>>(A, W, R, C, M, N, K);
  };

  // 1. instance norm + patchify (f16 A, K padded 16->32)
  k_instnorm<<<NB * NC, 256, 0, stream>>>(x_enc, meanv, stdv, Ap);

  // 2. weight conversions (f32 -> f16, zero-padded)
  cvt(patch_w, Wp, DM, DM, 16, 32);
  cvt(in_proj_w,                 Win0, 2 * DI, 2 * DI, DM, DM);
  cvt(in_proj_w + 2 * DI * DM,   Win1, 2 * DI, 2 * DI, DM, DM);
  cvt(x_proj_w,                  Wxp0, XDB, 40, DI, DI);
  cvt(x_proj_w + 40 * DI,        Wxp1, XDB, 40, DI, DI);
  cvt(out_proj_w,                Wout0, DM, DM, DI, DI);
  cvt(out_proj_w + DM * DI,      Wout1, DM, DM, DI, DI);
  cvt(head_w, Whead, PRED, PRED, KHEAD, KHEAD);

  // 3. patch embedding: x = Ap(16384x32) * Wp(128x32)^T
  gemm(Ap, Wp, nullptr, x, NTOK, DM, 32);

  // 4. two Mamba layers
  const _Float16* Win[2]  = { Win0, Win1 };
  const _Float16* Wxp[2]  = { Wxp0, Wxp1 };
  const _Float16* Wout[2] = { Wout0, Wout1 };
  for (int l = 0; l < 2; l++) {
    k_rmsnorm<<<NTOK, DM, 0, stream>>>(x, norm_w + l * DM, h16);
    gemm(h16, Win[l], nullptr, xz, NTOK, 2 * DI, DM);
    k_conv_silu<<<(NTOK * DI) / 256, 256, 0, stream>>>(
        xz, conv_w + l * DI * 4, conv_b + l * DI, u, u16);
    gemm(u16, Wxp[l], nullptr, xdbl, NTOK, XDB, DI);
    k_dt<<<(NTOK * DI) / 256, 256, 0, stream>>>(
        xdbl, dt_proj_w + l * DI * DTR, dt_proj_b + l * DI, dtb);
    k_scan<<<NB, DI, 0, stream>>>(
        dtb, u, xdbl, xz, A_log + l * DI * NST, D_skip + l * DI, y16);
    gemm(y16, Wout[l], /*R=*/x, /*C=*/x, NTOK, DM, DI);  // residual add
  }

  // 5. final norm + head GEMM: (b,c) rows of K=4096, N=96
  k_rmsnorm<<<NTOK, DM, 0, stream>>>(x, final_norm_w, h16);
  gemm(h16, Whead, nullptr, ho, NB * NC, PRED, KHEAD);

  // 6. bias + transpose + de-normalize
  k_head_out<<<(NB * PRED * NC + 255) / 256, 256, 0, stream>>>(
      ho, head_b, meanv, stdv, outp);
}
